// MixtureOfExperts_24103356465244
// MI455X (gfx1250) — matrix operand order repaired and checked by reading, not compiled, not run on hardware
//
#include <hip/hip_runtime.h>

// ---------------- problem constants (from reference setup_inputs) -------------
#define NTOK   8192          // B*T = 4*2048
#define DDIM   1024          // D
#define NEXP   8             // E
#define HDIM   2730          // H
#define HP     2816          // H padded to multiple of HC (22*128)
#define HC     128           // H chunk processed per iteration
#define NCHUNK (HP / HC)     // 22
#define MT     64            // tokens per expert tile (4x16 WMMA row blocks)

typedef unsigned short bfu;  // raw bf16 bits

typedef __attribute__((ext_vector_type(16))) __bf16 v16bf;
typedef __attribute__((ext_vector_type(8)))  float  v8f;
typedef __attribute__((ext_vector_type(4)))  int    v4i;

union V32 {            // 32 bytes: one WMMA bf16 operand (16 values/lane)
    v4i   i4[2];
    v16bf bf;
};

__device__ __forceinline__ bfu f32_to_bf16(float f) {
    unsigned int u = __float_as_uint(f);
    unsigned int lsb = (u >> 16) & 1u;
    u += 0x7fffu + lsb;               // round-to-nearest-even
    return (bfu)(u >> 16);
}

__device__ __forceinline__ unsigned int pack2_bf16(float a, float b) {
    return (unsigned int)f32_to_bf16(a) | ((unsigned int)f32_to_bf16(b) << 16);
}

// A operand: rowbase pre-offset by the lane's K-half (hi*16); 16 contiguous K values
__device__ __forceinline__ void ldA(V32& v, const bfu* rowbase, int k) {
    v.i4[0] = *(const v4i*)(rowbase + k);
    v.i4[1] = *(const v4i*)(rowbase + k + 8);
}
// B operand: 32 contiguous bytes (16 bf16 K-values for this lane's column)
__device__ __forceinline__ void ldB(V32& v, const bfu* p, int k) {
    v.i4[0] = ((const v4i*)(p + k))[0];
    v.i4[1] = ((const v4i*)(p + k))[1];
}

// ---------------------------------------------------------------------------
// Kernel 1: router. One wave per token: logits = x_row . gate_w[e], top-2,
// weights = softmax over the two selected logits (== renormalized top-k).
// ---------------------------------------------------------------------------
__global__ void moe_router(const float* __restrict__ x,
                           const float* __restrict__ gate_w,
                           int* __restrict__ tok_e0, int* __restrict__ tok_e1,
                           float* __restrict__ tok_w0, float* __restrict__ tok_w1) {
    const int tok  = (blockIdx.x * blockDim.x + threadIdx.x) >> 5;
    const int lane = threadIdx.x & 31;
    if (tok >= NTOK) return;

    const float* xr = x + (size_t)tok * DDIM;
    float xv[32];
#pragma unroll
    for (int j = 0; j < 32; ++j) xv[j] = xr[lane + 32 * j];

    float logit[NEXP];
#pragma unroll
    for (int e = 0; e < NEXP; ++e) {
        const float* gr = gate_w + (size_t)e * DDIM;
        float acc = 0.f;
#pragma unroll
        for (int j = 0; j < 32; ++j) acc += xv[j] * gr[lane + 32 * j];
#pragma unroll
        for (int off = 16; off > 0; off >>= 1) acc += __shfl_xor(acc, off, 32);
        logit[e] = acc;
    }
    if (lane == 0) {
        int i0 = 0; float l0 = logit[0];
#pragma unroll
        for (int e = 1; e < NEXP; ++e) if (logit[e] > l0) { l0 = logit[e]; i0 = e; }
        int i1 = -1; float l1 = -3.4e38f;
#pragma unroll
        for (int e = 0; e < NEXP; ++e) if (e != i0 && logit[e] > l1) { l1 = logit[e]; i1 = e; }
        float w0 = __builtin_amdgcn_rcpf(1.f + __expf(l1 - l0)); // exp(l0)/(exp(l0)+exp(l1))
        tok_e0[tok] = i0; tok_e1[tok] = i1;
        tok_w0[tok] = w0; tok_w1[tok] = 1.f - w0;
    }
}

// ---------------------------------------------------------------------------
// Kernel 2: deterministic ordered compaction of token ids per expert.
// One block per expert; LDS Hillis-Steele scan over 256-token chunks.
// ---------------------------------------------------------------------------
__global__ void moe_compact(const int* __restrict__ tok_e0, const int* __restrict__ tok_e1,
                            const float* __restrict__ tok_w0, const float* __restrict__ tok_w1,
                            int* __restrict__ tok_list, float* __restrict__ tok_wgt,
                            int* __restrict__ tok_slot, int* __restrict__ counts) {
    __shared__ int scan[256];
    const int e = blockIdx.x;
    const int t = threadIdx.x;
    int total = 0;
    for (int base = 0; base < NTOK; base += 256) {
        const int tok = base + t;
        const int s0 = (tok_e0[tok] == e);
        const int s1 = (tok_e1[tok] == e);
        const int sel = s0 | s1;                 // top-2 indices are distinct
        scan[t] = sel;
        __syncthreads();
        for (int off = 1; off < 256; off <<= 1) {
            int v = scan[t];
            int u = (t >= off) ? scan[t - off] : 0;
            __syncthreads();
            scan[t] = v + u;
            __syncthreads();
        }
        if (sel) {
            int pos = total + scan[t] - 1;
            tok_list[e * NTOK + pos] = tok;
            tok_wgt [e * NTOK + pos] = s0 ? tok_w0[tok] : tok_w1[tok];
            tok_slot[e * NTOK + pos] = s0 ? 0 : 1;
        }
        total += scan[255];
        __syncthreads();
    }
    if (t == 0) counts[e] = total;
}

// ---------------------------------------------------------------------------
// Kernels 3-5: vectorized f32 -> bf16 casts (4 elements / thread). Weights are
// zero-padded from H to HP so the fused kernel needs no tail handling
// (pad rows give silu(0)*0 = 0). 4-groups never straddle rows (D%4==HP%4==0).
// ---------------------------------------------------------------------------
__global__ void cast_x_bf16(const float* __restrict__ src, bfu* __restrict__ dst) {
    size_t i = ((size_t)blockIdx.x * blockDim.x + threadIdx.x) * 4;  // < NTOK*DDIM
    float4 v = *(const float4*)(src + i);
    uint2 r; r.x = pack2_bf16(v.x, v.y); r.y = pack2_bf16(v.z, v.w);
    *(uint2*)(dst + i) = r;
}

__global__ void cast_w13_bf16(const float* __restrict__ w, bfu* __restrict__ wb) {
    size_t i = ((size_t)blockIdx.x * blockDim.x + threadIdx.x) * 4;  // < NEXP*HP*DDIM
    int k = (int)(i % DDIM);
    int h = (int)((i / DDIM) % HP);
    int e = (int)(i / ((size_t)DDIM * HP));
    float4 v = make_float4(0.f, 0.f, 0.f, 0.f);
    if (h < HDIM) v = *(const float4*)(w + ((size_t)e * HDIM + h) * DDIM + k);
    uint2 r; r.x = pack2_bf16(v.x, v.y); r.y = pack2_bf16(v.z, v.w);
    *(uint2*)(wb + i) = r;
}

__global__ void cast_w2_bf16(const float* __restrict__ w, bfu* __restrict__ wb) {
    size_t i = ((size_t)blockIdx.x * blockDim.x + threadIdx.x) * 4;  // < NEXP*DDIM*HP
    int h = (int)(i % HP);
    int d = (int)((i / HP) % DDIM);
    int e = (int)(i / ((size_t)HP * DDIM));
    // h..h+3 may straddle the H boundary; handle per element
    const float* srow = w + ((size_t)e * DDIM + d) * HDIM;
    float v0 = (h + 0 < HDIM) ? srow[h + 0] : 0.f;
    float v1 = (h + 1 < HDIM) ? srow[h + 1] : 0.f;
    float v2 = (h + 2 < HDIM) ? srow[h + 2] : 0.f;
    float v3 = (h + 3 < HDIM) ? srow[h + 3] : 0.f;
    uint2 r; r.x = pack2_bf16(v0, v1); r.y = pack2_bf16(v2, v3);
    *(uint2*)(wb + i) = r;
}

// ---------------------------------------------------------------------------
// Kernel 6: fused SwiGLU expert GEMM. One workgroup (32 waves, wave32) handles
// MT=64 gathered tokens of one expert; the 64x1024 bf16 x-tile stays in LDS
// for the whole kernel. H processed in 128-wide chunks:
//   stage A: each wave computes one 16x16 act tile (silu(x@w1^T)*(x@w3^T))
//   stage B: each wave accumulates a 16x128 slice of out += act @ w2^T
// Operands are explicitly double-buffered so next-step loads overlap WMMA.
// ---------------------------------------------------------------------------
__launch_bounds__(1024, 1)
__global__ void moe_expert(const bfu* __restrict__ xb,
                           const bfu* __restrict__ w1b,
                           const bfu* __restrict__ w3b,
                           const bfu* __restrict__ w2b,
                           const int* __restrict__ tok_list,
                           const float* __restrict__ tok_wgt,
                           const int* __restrict__ tok_slot,
                           const int* __restrict__ counts,
                           float* __restrict__ y_tmp) {
    __shared__ bfu   xs[MT][DDIM + 8];   // 64 token rows, bf16, padded pitch (132KB)
    __shared__ bfu   act[MT][HC + 8];    // current activation chunk (17.4KB)
    __shared__ int   toksh[MT];
    __shared__ float wgtsh[MT];
    __shared__ int   slotsh[MT];

    const int e    = blockIdx.y;
    const int tile = blockIdx.x;
    const int cnt  = counts[e];
    if (tile * MT >= cnt) return;        // uniform early-exit per block

    const int tid  = threadIdx.x;
    const int wv   = tid >> 5;           // 0..31
    const int lane = tid & 31;
    const int ln16 = lane & 15;
    const int hi   = lane >> 4;          // 0: K-half {0..15}; 1: {16..31}
    const int msub = wv >> 3;            // 0..3 : 16-token row block
    const int nsl  = wv & 7;             // 0..7 : 16-wide column slot

    if (tid < MT) {
        int pos = tile * MT + tid;
        int ok  = pos < cnt;
        toksh[tid]  = ok ? tok_list[e * NTOK + pos] : -1;
        wgtsh[tid]  = ok ? tok_wgt [e * NTOK + pos] : 0.f;
        slotsh[tid] = ok ? tok_slot[e * NTOK + pos] : 0;
    }
    __syncthreads();

    // Gather MT token rows (bf16) into LDS; 16 threads per row, 128B each.
    {
        int row  = tid & (MT - 1);
        int cseg = tid >> 6;                      // 0..15, 64 elements each
        int tok  = toksh[row]; if (tok < 0) tok = 0;
        const bfu* src = xb + (size_t)tok * DDIM + cseg * 64;
#pragma unroll
        for (int j = 0; j < 8; ++j)
            *((v4i*)&xs[row][cseg * 64 + j * 8]) = ((const v4i*)src)[j];
    }
    __syncthreads();

    v8f acc[8];                                   // wave-private 16x128 out slice
#pragma unroll
    for (int t = 0; t < 8; ++t)
#pragma unroll
        for (int i = 0; i < 8; ++i) acc[t][i] = 0.f;

    // persistent operand pointers (advanced per chunk)
    const bfu* xrow  = &xs[msub * 16 + ln16][hi * 16];      // stage-A A operand base
    const bfu* arow  = &act[msub * 16 + ln16][hi * 16];     // stage-B A operand base
    const bfu* w1p = w1b + (size_t)e * HP * DDIM + (size_t)(nsl * 16 + ln16) * DDIM + hi * 16;
    const bfu* w3p = w3b + (size_t)e * HP * DDIM + (size_t)(nsl * 16 + ln16) * DDIM + hi * 16;
    const bfu* bp[8];
#pragma unroll
    for (int t = 0; t < 8; ++t)
        bp[t] = w2b + ((size_t)e * DDIM + nsl * 128 + t * 16 + ln16) * HP + hi * 16;

    for (int c = 0; c < NCHUNK; ++c) {
        if (c + 1 < NCHUNK) {                     // next-chunk weight prefetch
            __builtin_prefetch(w1p + (size_t)HC * DDIM, 0, 1);
            __builtin_prefetch(w3p + (size_t)HC * DDIM, 0, 1);
        }

        // ---- stage A: 16x16 act tile at h rows [c*HC + nsl*16, +16), K over D.
        v8f g, u;
#pragma unroll
        for (int i = 0; i < 8; ++i) { g[i] = 0.f; u[i] = 0.f; }

        {
            V32 aa[2], b1v[2], b3v[2];
            ldA(aa[0], xrow, 0);
            ldB(b1v[0], w1p, 0);
            ldB(b3v[0], w3p, 0);
#pragma unroll
            for (int ks = 0; ks < DDIM / 32; ++ks) {           // 32 k-steps
                if (ks + 1 < DDIM / 32) {                      // prefetch next step
                    ldA(aa[(ks + 1) & 1], xrow, (ks + 1) * 32);
                    ldB(b1v[(ks + 1) & 1], w1p, (ks + 1) * 32);
                    ldB(b3v[(ks + 1) & 1], w3p, (ks + 1) * 32);
                }
                g = __builtin_amdgcn_wmma_f32_16x16x32_bf16(false, aa[ks & 1].bf, false,
                                                            b1v[ks & 1].bf, (short)0, g,
                                                            false, false);
                u = __builtin_amdgcn_wmma_f32_16x16x32_bf16(false, aa[ks & 1].bf, false,
                                                            b3v[ks & 1].bf, (short)0, u,
                                                            false, false);
            }
        }

        __syncthreads();   // previous chunk's stage-B act reads complete
#pragma unroll
        for (int i = 0; i < 8; ++i) {
            float gv = g[i];
            float sv = gv * __builtin_amdgcn_rcpf(1.f + __expf(-gv));   // fast silu
            act[msub * 16 + i + (hi << 3)][nsl * 16 + ln16] = f32_to_bf16(sv * u[i]);
        }
        __syncthreads();   // act chunk visible to all waves

        // ---- stage B: out rows [msub*16,+16) x cols [nsl*128,+128) += act @ w2^T
        {
            V32 ab[2];
            ldA(ab[0], arow, 0);
#pragma unroll
            for (int ks = 0; ks < HC / 32; ++ks) {             // 4 k-steps
                if (ks + 1 < HC / 32) ldA(ab[(ks + 1) & 1], arow, (ks + 1) * 32);
                V32 bb[2];
                ldB(bb[0], bp[0], ks * 32);
#pragma unroll
                for (int t = 0; t < 8; ++t) {
                    if (t + 1 < 8) ldB(bb[(t + 1) & 1], bp[t + 1], ks * 32);
                    acc[t] = __builtin_amdgcn_wmma_f32_16x16x32_bf16(false, ab[ks & 1].bf,
                                                                     false, bb[t & 1].bf,
                                                                     (short)0, acc[t],
                                                                     false, false);
                }
            }
        }
        // barrier before next chunk's act store is the __syncthreads above

        // advance weight pointers to next H chunk
        w1p += (size_t)HC * DDIM;
        w3p += (size_t)HC * DDIM;
#pragma unroll
        for (int t = 0; t < 8; ++t) bp[t] += HC;
    }

    // ---- epilogue: gated write into per-(token,slot) buffer (no atomics)
#pragma unroll
    for (int i = 0; i < 8; ++i) {
        const int row = msub * 16 + i + (hi << 3);
        const int tok = toksh[row];
        if (tok >= 0) {
            const float w = wgtsh[row];
            const size_t base = ((size_t)tok * 2 + slotsh[row]) * DDIM + nsl * 128 + ln16;
#pragma unroll
            for (int t = 0; t < 8; ++t) y_tmp[base + t * 16] = w * acc[t][i];
        }
    }
}

// ---------------------------------------------------------------------------
// Kernel 7: deterministic combine of the two slots + aux_loss.
// ---------------------------------------------------------------------------
__global__ void moe_combine(const float* __restrict__ y_tmp, float* __restrict__ out) {
    size_t idx = (size_t)blockIdx.x * blockDim.x + threadIdx.x;  // < NTOK*DDIM
    size_t tok = idx / DDIM;
    size_t d   = idx - tok * DDIM;
    out[idx] = y_tmp[tok * 2 * DDIM + d] + y_tmp[(tok * 2 + 1) * DDIM + d];
    if (idx == 0) out[(size_t)NTOK * DDIM] = 0.f;                // aux_loss
}

// ---------------------------------------------------------------------------
extern "C" void kernel_launch(void* const* d_in, const int* in_sizes, int n_in,
                              void* d_out, int out_size, void* d_ws, size_t ws_size,
                              hipStream_t stream) {
    const float* x      = (const float*)d_in[0];
    const float* gate_w = (const float*)d_in[1];
    const float* w1     = (const float*)d_in[2];
    const float* w3     = (const float*)d_in[3];
    const float* w2     = (const float*)d_in[4];
    float* out = (float*)d_out;
    (void)in_sizes; (void)n_in; (void)out_size; (void)ws_size;

    // workspace layout (all 256B aligned)
    char* ws = (char*)d_ws;
    size_t off = 0;
    auto take = [&](size_t bytes) -> void* {
        void* p = ws + off;
        off += (bytes + 255) & ~(size_t)255;
        return p;
    };
    bfu*   xb       = (bfu*)  take((size_t)NTOK * DDIM * 2);
    bfu*   w1b      = (bfu*)  take((size_t)NEXP * HP * DDIM * 2);
    bfu*   w3b      = (bfu*)  take((size_t)NEXP * HP * DDIM * 2);
    bfu*   w2b      = (bfu*)  take((size_t)NEXP * DDIM * HP * 2);
    float* y_tmp    = (float*)take((size_t)NTOK * 2 * DDIM * 4);
    int*   tok_e0   = (int*)  take((size_t)NTOK * 4);
    int*   tok_e1   = (int*)  take((size_t)NTOK * 4);
    float* tok_w0   = (float*)take((size_t)NTOK * 4);
    float* tok_w1   = (float*)take((size_t)NTOK * 4);
    int*   tok_list = (int*)  take((size_t)NEXP * NTOK * 4);
    float* tok_wgt  = (float*)take((size_t)NEXP * NTOK * 4);
    int*   tok_slot = (int*)  take((size_t)NEXP * NTOK * 4);
    int*   counts   = (int*)  take((size_t)NEXP * 4);

    // 1) router: 8 waves/block -> 8 tokens per block
    moe_router<<<NTOK / 8, 256, 0, stream>>>(x, gate_w, tok_e0, tok_e1, tok_w0, tok_w1);
    // 2) deterministic per-expert compaction
    moe_compact<<<NEXP, 256, 0, stream>>>(tok_e0, tok_e1, tok_w0, tok_w1,
                                          tok_list, tok_wgt, tok_slot, counts);
    // 3-5) bf16 casts (weights padded H -> HP), 4 elements per thread
    cast_x_bf16 <<<(NTOK * DDIM) / (256 * 4), 256, 0, stream>>>(x, xb);
    cast_w13_bf16<<<((size_t)NEXP * HP * DDIM) / (256 * 4), 256, 0, stream>>>(w1, w1b);
    cast_w13_bf16<<<((size_t)NEXP * HP * DDIM) / (256 * 4), 256, 0, stream>>>(w3, w3b);
    cast_w2_bf16 <<<((size_t)NEXP * DDIM * HP) / (256 * 4), 256, 0, stream>>>(w2, w2b);
    // 6) fused expert GEMMs: tiles x experts; empty tiles exit via counts[]
    moe_expert<<<dim3(NTOK / MT, NEXP), 1024, 0, stream>>>(xb, w1b, w3b, w2b,
                                                           tok_list, tok_wgt, tok_slot,
                                                           counts, y_tmp);
    // 7) combine slots + aux loss
    moe_combine<<<(NTOK * DDIM) / 256, 256, 0, stream>>>(y_tmp, out);
}
